// MambaBlock_6897717477938
// MI455X (gfx1250) — matrix operand order repaired and checked by reading, not compiled, not run on hardware
//
#include <hip/hip_runtime.h>
#include <hip/hip_bf16.h>

// ---------------------------------------------------------------------------
// Mamba2 block for MI455X (gfx1250). GEMMs run on v_wmma_f32_16x16x32_bf16
// (bf16 inputs, f32 accumulate) with a 32x32 per-wave tile and double-buffered
// global loads (branchless inner loop, peeled tail); scan/conv/norms in f32.
// ---------------------------------------------------------------------------

typedef __attribute__((ext_vector_type(16))) __bf16 v16bf;
typedef __attribute__((ext_vector_type(8)))  __bf16 v8bf;
typedef __attribute__((ext_vector_type(4)))  __bf16 v4bf;
typedef __attribute__((ext_vector_type(8)))  float  v8f;

#define L_SEQ   2048
#define D_MODEL 1024
#define D_INNER 2048
#define D_STATE 128
#define NHEADS  32
#define HEADDIM 64
#define CONV_DIM 2304
#define N_INPROJ 4384
#define INTER   2752

enum { EPI_F32 = 0, EPI_ADD_F32 = 1, EPI_SILU_GATE_BF16 = 2 };

__device__ __forceinline__ float silu_f(float x) { return x / (1.f + __expf(-x)); }

// --------------------------- f32 -> bf16 convert ---------------------------
__global__ __launch_bounds__(256) void cvt_bf16_x4(const float4* __restrict__ in,
                                                   __bf16* __restrict__ out, long n4) {
  long i = (long)blockIdx.x * 256 + threadIdx.x;
  if (i >= n4) return;
  float4 v = in[i];
  v4bf o; o.x = (__bf16)v.x; o.y = (__bf16)v.y; o.z = (__bf16)v.z; o.w = (__bf16)v.w;
  *(v4bf*)(out + 4 * i) = o;
}

// ------------------------------- RMSNorm -----------------------------------
__global__ __launch_bounds__(256) void rmsnorm_bf16(const float* __restrict__ x,
                                                    const float* __restrict__ w,
                                                    __bf16* __restrict__ out, int D) {
  const int row = blockIdx.x;
  const float* xr = x + (long)row * D;
  float ss = 0.f;
  for (int i = threadIdx.x; i < D; i += 256) { float v = xr[i]; ss += v * v; }
  #pragma unroll
  for (int off = 16; off; off >>= 1) ss += __shfl_xor(ss, off, 32);
  __shared__ float sbuf[8];
  if ((threadIdx.x & 31) == 0) sbuf[threadIdx.x >> 5] = ss;
  __syncthreads();
  float tot = 0.f;
  #pragma unroll
  for (int i = 0; i < 8; ++i) tot += sbuf[i];
  const float scale = rsqrtf(tot / (float)D + 1e-6f);
  for (int i = threadIdx.x; i < D; i += 256)
    out[(long)row * D + i] = (__bf16)(xr[i] * scale * w[i]);
}

// ------------------- gated RMSNorm: rmsnorm(y*silu(z)) ---------------------
__global__ __launch_bounds__(256) void gated_rmsnorm(const float* __restrict__ y,
                                                     const float* __restrict__ zx,
                                                     const float* __restrict__ w,
                                                     __bf16* __restrict__ out) {
  const int row = blockIdx.x;
  const float* yr = y + (long)row * D_INNER;
  const float* zr = zx + (long)row * N_INPROJ;   // z = cols [0, 2048)
  float gv[8];
  float ss = 0.f;
  #pragma unroll
  for (int j = 0; j < 8; ++j) {
    int i = threadIdx.x + j * 256;
    float g = yr[i] * silu_f(zr[i]);
    gv[j] = g; ss += g * g;
  }
  #pragma unroll
  for (int off = 16; off; off >>= 1) ss += __shfl_xor(ss, off, 32);
  __shared__ float sbuf[8];
  if ((threadIdx.x & 31) == 0) sbuf[threadIdx.x >> 5] = ss;
  __syncthreads();
  float tot = 0.f;
  #pragma unroll
  for (int i = 0; i < 8; ++i) tot += sbuf[i];
  const float scale = rsqrtf(tot / (float)D_INNER + 1e-6f);
  #pragma unroll
  for (int j = 0; j < 8; ++j) {
    int i = threadIdx.x + j * 256;
    out[(long)row * D_INNER + i] = (__bf16)(gv[j] * scale * w[i]);
  }
}

// ------------------- causal depthwise conv (K=4) + SiLU --------------------
__global__ __launch_bounds__(256) void conv_silu(const float* __restrict__ zx,
                                                 const float* __restrict__ cw,
                                                 const float* __restrict__ cb,
                                                 float* __restrict__ xBC) {
  long idx = (long)blockIdx.x * 256 + threadIdx.x;
  if (idx >= (long)L_SEQ * CONV_DIM) return;
  int t = (int)(idx / CONV_DIM);
  int c = (int)(idx % CONV_DIM);
  float acc = cb[c];
  #pragma unroll
  for (int j = 0; j < 4; ++j) {
    int tt = t - 3 + j;
    if (tt >= 0) acc += cw[c * 4 + j] * zx[(long)tt * N_INPROJ + D_INNER + c];
  }
  xBC[idx] = silu_f(acc);
}

// ----------------------------- SSM scan ------------------------------------
// One block per head; 256 threads hold the 64x128 state in registers
// (32 floats/thread). B/C/x staged in LDS per timestep.
__global__ __launch_bounds__(256) void ssm_scan(const float* __restrict__ zx,
                                                const float* __restrict__ xBC,
                                                const float* __restrict__ dt_bias,
                                                const float* __restrict__ A_log,
                                                const float* __restrict__ Dp,
                                                float* __restrict__ y) {
  const int h   = blockIdx.x;
  const int tid = threadIdx.x;
  const int p   = tid >> 2;          // head-dim index 0..63
  const int nb  = (tid & 3) << 5;    // state base 0/32/64/96
  __shared__ float sB[D_STATE], sC[D_STATE], sx[HEADDIM];
  float s[32];
  #pragma unroll
  for (int i = 0; i < 32; ++i) s[i] = 0.f;
  const float A  = -__expf(A_log[h]);
  const float db = dt_bias[h];
  const float Dh = Dp[h];
  for (int t = 0; t < L_SEQ; ++t) {
    const float* xr = xBC + (long)t * CONV_DIM;
    if (tid < 128)  sB[tid] = xr[D_INNER + tid];
    if (tid >= 128) sC[tid - 128] = xr[D_INNER + D_STATE + (tid - 128)];
    if (tid < 64)   sx[tid] = xr[h * HEADDIM + tid];
    __syncthreads();
    float dtr = zx[(long)t * N_INPROJ + (D_INNER + CONV_DIM) + h] + db;
    float dt  = dtr > 20.f ? dtr : log1pf(__expf(dtr));   // softplus
    float dA  = __expf(dt * A);
    float xv  = sx[p];
    float coef = dt * xv;
    float ysum = 0.f;
    #pragma unroll
    for (int i = 0; i < 32; ++i) {
      s[i] = s[i] * dA + coef * sB[nb + i];
      ysum += s[i] * sC[nb + i];
    }
    ysum += __shfl_xor(ysum, 1, 32);
    ysum += __shfl_xor(ysum, 2, 32);
    if ((tid & 3) == 0)
      y[(long)t * D_INNER + h * HEADDIM + p] = ysum + Dh * xv;
    __syncthreads();
  }
}

// ------------------------ WMMA GEMM: C = A * B^T ---------------------------
// A: [M,K] bf16 row-major, B: [N,K] bf16 row-major (weights as (out,in)).
// One wave per 32x32 output tile (2 A-frags x 2 B-frags -> 4 accumulators,
// 4 v_wmma per K=32 step, 8 b128 loads). Loads double-buffered with a peeled
// tail so the inner loop is branchless; each WMMA group depends only on the
// previous iteration's loads.

// A fragment: lane m holds K koff..koff+7 (V0..3) and koff+16..+23 (V4..7)
__device__ __forceinline__ v16bf load_frag_a(const __bf16* __restrict__ p) {
  v8bf lo = *(const v8bf*)(p);
  v8bf hi = *(const v8bf*)(p + 16);
  v16bf r;
  #pragma unroll
  for (int i = 0; i < 8; ++i) { r[i] = lo[i]; r[i + 8] = hi[i]; }
  return r;
}
// B fragment: lane n holds K koff..koff+7 (V0..3) and koff+16..+23 (V4..7)
__device__ __forceinline__ v16bf load_frag_b(const __bf16* __restrict__ p) {
  v8bf lo = *(const v8bf*)(p);
  v8bf hi = *(const v8bf*)(p + 16);
  v16bf r;
  #pragma unroll
  for (int i = 0; i < 8; ++i) { r[i] = lo[i]; r[i + 8] = hi[i]; }
  return r;
}

__global__ __launch_bounds__(256) void gemm_bf16_wmma(const __bf16* __restrict__ A,
                                                      const __bf16* __restrict__ B,
                                                      void* __restrict__ out,
                                                      const float* __restrict__ aux,
                                                      int M, int N, int K, int epi) {
  const int lane = threadIdx.x & 31;
  const int wave = threadIdx.x >> 5;
  const int tile = blockIdx.x * 8 + wave;
  const int nTiles = N >> 5;                 // 32-wide N tiles
  const int mTile = tile / nTiles;
  const int nTile = tile - mTile * nTiles;
  if (mTile >= (M >> 5)) return;             // wave-uniform
  const int m0 = mTile << 5;
  const int n0 = nTile << 5;

  const int hi  = lane >> 4;
  const int l15 = lane & 15;
  // per-lane base pointers (K offsets added in-loop)
  const __bf16* pa0 = A + (long)(m0 + l15) * K + (hi << 3);
  const __bf16* pa1 = A + (long)(m0 + 16 + l15) * K + (hi << 3);
  const __bf16* pb0 = B + (long)(n0 + l15) * K + (hi << 3);
  const __bf16* pb1 = B + (long)(n0 + 16 + l15) * K + (hi << 3);

  v8f c00 = {}, c01 = {}, c10 = {}, c11 = {};

  v16bf a0 = load_frag_a(pa0);
  v16bf a1 = load_frag_a(pa1);
  v16bf b0 = load_frag_b(pb0);
  v16bf b1 = load_frag_b(pb1);

  // branchless pipelined main loop; final K-step peeled below
  for (int k = 32; k < K; k += 32) {
    v16bf na0 = load_frag_a(pa0 + k);
    v16bf na1 = load_frag_a(pa1 + k);
    v16bf nb0 = load_frag_b(pb0 + k);
    v16bf nb1 = load_frag_b(pb1 + k);
    __builtin_prefetch(pa0 + k + 32, 0, 1);   // global_prefetch_b8, L2 warm-up
    __builtin_prefetch(pb0 + k + 32, 0, 1);
    __builtin_prefetch(pb1 + k + 32, 0, 1);
    c00 = __builtin_amdgcn_wmma_f32_16x16x32_bf16(false, a0, false, b0, (short)0, c00, false, false);
    c01 = __builtin_amdgcn_wmma_f32_16x16x32_bf16(false, a0, false, b1, (short)0, c01, false, false);
    c10 = __builtin_amdgcn_wmma_f32_16x16x32_bf16(false, a1, false, b0, (short)0, c10, false, false);
    c11 = __builtin_amdgcn_wmma_f32_16x16x32_bf16(false, a1, false, b1, (short)0, c11, false, false);
    a0 = na0; a1 = na1; b0 = nb0; b1 = nb1;
  }
  c00 = __builtin_amdgcn_wmma_f32_16x16x32_bf16(false, a0, false, b0, (short)0, c00, false, false);
  c01 = __builtin_amdgcn_wmma_f32_16x16x32_bf16(false, a0, false, b1, (short)0, c01, false, false);
  c10 = __builtin_amdgcn_wmma_f32_16x16x32_bf16(false, a1, false, b0, (short)0, c10, false, false);
  c11 = __builtin_amdgcn_wmma_f32_16x16x32_bf16(false, a1, false, b1, (short)0, c11, false, false);

  // C/D layout: VGPR g -> row +g (lanes 0-15) or +g+8 (lanes 16-31); col = lane&15
  const int rb   = m0 + (hi << 3);
  const int col0 = n0 + l15;
  const int col1 = col0 + 16;
  if (epi == EPI_SILU_GATE_BF16) {
    __bf16* o = (__bf16*)out;
    #pragma unroll
    for (int g = 0; g < 8; ++g) {
      long r0 = (long)(rb + g) * N;
      long r1 = (long)(rb + 16 + g) * N;
      o[r0 + col0] = (__bf16)(silu_f(aux[r0 + col0]) * c00[g]);
      o[r0 + col1] = (__bf16)(silu_f(aux[r0 + col1]) * c01[g]);
      o[r1 + col0] = (__bf16)(silu_f(aux[r1 + col0]) * c10[g]);
      o[r1 + col1] = (__bf16)(silu_f(aux[r1 + col1]) * c11[g]);
    }
  } else {
    float* o = (float*)out;
    #pragma unroll
    for (int g = 0; g < 8; ++g) {
      long r0 = (long)(rb + g) * N;
      long r1 = (long)(rb + 16 + g) * N;
      float v00 = c00[g], v01 = c01[g], v10 = c10[g], v11 = c11[g];
      if (epi == EPI_ADD_F32) {
        v00 += aux[r0 + col0]; v01 += aux[r0 + col1];
        v10 += aux[r1 + col0]; v11 += aux[r1 + col1];
      }
      o[r0 + col0] = v00; o[r0 + col1] = v01;
      o[r1 + col0] = v10; o[r1 + col1] = v11;
    }
  }
}

// ---------------------------------------------------------------------------
extern "C" void kernel_launch(void* const* d_in, const int* in_sizes, int n_in,
                              void* d_out, int out_size, void* d_ws, size_t ws_size,
                              hipStream_t stream) {
  (void)in_sizes; (void)n_in; (void)out_size; (void)ws_size;
  const float* hidden     = (const float*)d_in[0];
  const float* norm_w     = (const float*)d_in[1];
  const float* in_proj_w  = (const float*)d_in[2];
  const float* conv_w     = (const float*)d_in[3];
  const float* conv_b     = (const float*)d_in[4];
  const float* dt_bias    = (const float*)d_in[5];
  const float* A_log      = (const float*)d_in[6];
  const float* Dp         = (const float*)d_in[7];
  const float* ssm_norm_w = (const float*)d_in[8];
  const float* out_proj_w = (const float*)d_in[9];
  const float* post_norm_w= (const float*)d_in[10];
  const float* gate_w     = (const float*)d_in[11];
  const float* up_w       = (const float*)d_in[12];
  const float* down_w     = (const float*)d_in[13];
  float* out = (float*)d_out;

  char* base = (char*)d_ws;
  size_t off = 0;
  auto take = [&](size_t bytes) -> char* {
    char* p = base + off;
    off = (off + bytes + 255) & ~(size_t)255;
    return p;
  };
  // bf16 weight copies (persist through launch)
  __bf16* wi = (__bf16*)take((size_t)N_INPROJ * D_MODEL * 2);
  __bf16* wo = (__bf16*)take((size_t)D_MODEL * D_INNER * 2);
  __bf16* wg = (__bf16*)take((size_t)INTER * D_MODEL * 2);
  __bf16* wu = (__bf16*)take((size_t)INTER * D_MODEL * 2);
  __bf16* wd = (__bf16*)take((size_t)D_MODEL * INTER * 2);
  // activations (time-aliased)
  __bf16* xnorm = (__bf16*)take((size_t)L_SEQ * D_MODEL * 2);   // later reused as h2n
  float*  zx    = (float*) take((size_t)L_SEQ * N_INPROJ * 4);  // later reused as gate_f
  float*  xBC   = (float*) take((size_t)L_SEQ * CONV_DIM * 4);  // later reused as gu (bf16)
  float*  yf    = (float*) take((size_t)L_SEQ * D_INNER * 4);
  __bf16* ybf   = (__bf16*)take((size_t)L_SEQ * D_INNER * 2);
  float*  h2    = (float*) take((size_t)L_SEQ * D_MODEL * 4);
  __bf16* h2n   = xnorm;                  // xnorm dead after in_proj GEMM
  float*  gate_f= zx;                     // zxbcdt dead after gated_rmsnorm
  __bf16* gu    = (__bf16*)xBC;           // xBC dead after ssm_scan

  auto cvt = [&](const float* src, __bf16* dst, long n) {
    long n4 = n / 4;
    cvt_bf16_x4<<<(unsigned)((n4 + 255) / 256), 256, 0, stream>>>((const float4*)src, dst, n4);
  };
  auto gemm = [&](const __bf16* A, const __bf16* B, void* o, const float* aux,
                  int M, int N, int K, int epi) {
    int tiles = (M / 32) * (N / 32);
    gemm_bf16_wmma<<<(tiles + 7) / 8, 256, 0, stream>>>(A, B, o, aux, M, N, K, epi);
  };

  cvt(in_proj_w,  wi, (long)N_INPROJ * D_MODEL);
  cvt(out_proj_w, wo, (long)D_MODEL * D_INNER);
  cvt(gate_w,     wg, (long)INTER * D_MODEL);
  cvt(up_w,       wu, (long)INTER * D_MODEL);
  cvt(down_w,     wd, (long)D_MODEL * INTER);

  // h = rmsnorm(hidden) -> bf16
  rmsnorm_bf16<<<L_SEQ, 256, 0, stream>>>(hidden, norm_w, xnorm, D_MODEL);
  // zxbcdt = h @ in_proj_w^T
  gemm(xnorm, wi, zx, nullptr, L_SEQ, N_INPROJ, D_MODEL, EPI_F32);
  // xBC = silu(causal_conv(zxbcdt[:,2048:4352]) + b)
  conv_silu<<<(unsigned)(((long)L_SEQ * CONV_DIM + 255) / 256), 256, 0, stream>>>(zx, conv_w, conv_b, xBC);
  // sequential selective scan, y += D*x
  ssm_scan<<<NHEADS, 256, 0, stream>>>(zx, xBC, dt_bias, A_log, Dp, yf);
  // ybf = rmsnorm(y * silu(z))
  gated_rmsnorm<<<L_SEQ, 256, 0, stream>>>(yf, zx, ssm_norm_w, ybf);
  // h2 = hidden + ybf @ out_proj_w^T
  gemm(ybf, wo, h2, hidden, L_SEQ, D_MODEL, D_INNER, EPI_ADD_F32);
  // h2n = rmsnorm(h2)
  rmsnorm_bf16<<<L_SEQ, 256, 0, stream>>>(h2, post_norm_w, h2n, D_MODEL);
  // MLP: gate, fused silu(gate)*up -> bf16, down + residual
  gemm(h2n, wg, gate_f, nullptr, L_SEQ, INTER, D_MODEL, EPI_F32);
  gemm(h2n, wu, gu, gate_f, L_SEQ, INTER, D_MODEL, EPI_SILU_GATE_BF16);
  gemm(gu, wd, out, h2, L_SEQ, D_MODEL, INTER, EPI_ADD_F32);
}
